// ContrastiveLoss_15126874816925
// MI455X (gfx1250) — matrix operand order repaired and checked by reading, not compiled, not run on hardware
//
#include <hip/hip_runtime.h>
#include <stdint.h>

// ---------------------------------------------------------------------------
// ContrastiveLoss (cross-attention t2i) for MI455X / gfx1250.
// B=128, Li=36 image regions, Lw=50 caption words, D=1024.
// Heavy op: A[j,i,r,w] = <im[i,r,:], s[j,w,:]>  -> f16 WMMA (f32 accum),
// fed by double-buffered async global->LDS copies (ASYNCcnt pipeline).
// ---------------------------------------------------------------------------

typedef __attribute__((ext_vector_type(16))) _Float16 v16h;
typedef __attribute__((ext_vector_type(8)))  _Float16 v8h;
typedef __attribute__((ext_vector_type(8)))  float    v8f;

#define NB   128
#define LI   36
#define LW   50
#define DD   1024
#define LAMBDA_SM 9.0f
#define MARGIN 0.2f
#define EPSV 1e-8f

#define A_ROWS 48            // 3 M-tiles of 16 (rows >=36 are don't-care)
#define B_ROWS 64            // 4 N-tiles of 16 (rows >=50 are don't-care)
#define STG    72            // f16 stride (144B) -> bank-friendly b128 reads
#define A_TILE_BYTES (A_ROWS * STG * 2)   // 6912
#define B_TILE_BYTES (B_ROWS * STG * 2)   // 9216
#define NCHUNK (DD / 64)     // 16 K-chunks of 64

// ---------------------------------------------------------------- convert f32 -> f16
__global__ void cvt_f16_kernel(const float* __restrict__ src,
                               _Float16* __restrict__ dst, int n) {
    int t = blockIdx.x * blockDim.x + threadIdx.x;
    if (t < n) dst[t] = (_Float16)src[t];
}

// ---------------------------------------------------------------- caption word norms
__global__ void capnorm_kernel(const float* __restrict__ s,
                               float* __restrict__ capn) {
    int t = blockIdx.x * blockDim.x + threadIdx.x;  // t in [0, NB*LW)
    if (t < NB * LW) {
        const float* row = s + (size_t)t * DD;
        float acc = 0.f;
        for (int k = 0; k < DD; ++k) acc += row[k] * row[k];
        capn[t] = sqrtf(acc);
    }
}

// ---------------------------------------------------------------- per-image Gram G = im @ im^T
__global__ void gram_kernel(const float* __restrict__ im,
                            float* __restrict__ G) {
    int i = blockIdx.x;
    const float* base = im + (size_t)i * LI * DD;
    for (int e = threadIdx.x; e < LI * LI; e += blockDim.x) {
        int r = e / LI, c = e % LI;
        const float* a = base + (size_t)r * DD;
        const float* b = base + (size_t)c * DD;
        float acc = 0.f;
        for (int k = 0; k < DD; ++k) acc += a[k] * b[k];
        G[(size_t)i * LI * LI + e] = acc;
    }
}

// ---------------------------------------------------------------- main pair kernel
// block = (caption j, image i), 128 threads = 4 waves; wave w owns word cols 16w..16w+15
__global__ __launch_bounds__(128)
void pair_kernel(const _Float16* __restrict__ imh,
                 const _Float16* __restrict__ sh,
                 const float* __restrict__ capn,
                 const float* __restrict__ Gg,
                 const int* __restrict__ s_l,
                 float* __restrict__ scores) {
    const int j    = blockIdx.x;           // caption
    const int i    = blockIdx.y;           // image
    const int tid  = threadIdx.x;
    const int lane = tid & 31;
    const int wv   = tid >> 5;             // wave id = N-tile index
    const int half = lane >> 4;
    const int l16  = lane & 15;
    const int nw   = s_l[j];

    __shared__ __align__(16) _Float16 a_stage[2 * A_ROWS * STG];
    __shared__ __align__(16) _Float16 b_stage[2 * B_ROWS * STG];
    __shared__ float A_lds[LI * 65];                      // A[r][w], stride 65
    __shared__ float W_lds[LI * 65];                      // a1 then attn, stride 65
    __shared__ float G_lds[LI * LI];
    __shared__ float red[128];

    for (int t = tid; t < LI * LI; t += 128)
        G_lds[t] = Gg[(size_t)i * LI * LI + t];

    const _Float16* imrow = imh + (size_t)i * LI * DD;
    const _Float16* srow  = sh  + (size_t)j * LW * DD;

    // --- per-thread async-copy slot precompute (7 b128 copies per chunk) ---
    // Rows beyond the real matrices are CLAMPED (not skipped): garbage in
    // A-rows>=36 / B-rows>=50 only affects WMMA outputs we never read, and
    // keeps the per-wave ASYNCcnt an exact constant for the pipeline waits.
    const uint32_t aLds0 = (uint32_t)(size_t)a_stage;
    const uint32_t bLds0 = (uint32_t)(size_t)b_stage;
    uint32_t aGoff[3], aLoff[3], bGoff[4], bLoff[4];
    #pragma unroll
    for (int q = 0; q < 3; ++q) {
        int slot = tid + q * 128;                 // < 384
        int r = slot >> 3, cb = (slot & 7) * 8;
        int rc = r < LI ? r : (LI - 1);
        aGoff[q] = (uint32_t)((rc * DD + cb) * 2);
        aLoff[q] = (uint32_t)((r * STG + cb) * 2);
    }
    #pragma unroll
    for (int q = 0; q < 4; ++q) {
        int slot = tid + q * 128;                 // < 512
        int r = slot >> 3, cb = (slot & 7) * 8;
        int rc = r < LW ? r : (LW - 1);
        bGoff[q] = (uint32_t)((rc * DD + cb) * 2);
        bLoff[q] = (uint32_t)((r * STG + cb) * 2);
    }

#define ISSUE_CHUNK(k0bytes, bufsel)                                         \
    do {                                                                     \
        uint32_t ab_ = aLds0 + (uint32_t)(bufsel) * A_TILE_BYTES;            \
        uint32_t bb_ = bLds0 + (uint32_t)(bufsel) * B_TILE_BYTES;            \
        uint32_t kb_ = (uint32_t)(k0bytes);                                  \
        _Pragma("unroll")                                                    \
        for (int q_ = 0; q_ < 3; ++q_) {                                     \
            uint32_t l_ = ab_ + aLoff[q_];                                   \
            uint32_t g_ = aGoff[q_] + kb_;                                   \
            asm volatile("global_load_async_to_lds_b128 %0, %1, %2"          \
                         :: "v"(l_), "v"(g_), "s"(imrow) : "memory");        \
        }                                                                    \
        _Pragma("unroll")                                                    \
        for (int q_ = 0; q_ < 4; ++q_) {                                     \
            uint32_t l_ = bb_ + bLoff[q_];                                   \
            uint32_t g_ = bGoff[q_] + kb_;                                   \
            asm volatile("global_load_async_to_lds_b128 %0, %1, %2"          \
                         :: "v"(l_), "v"(g_), "s"(srow) : "memory");         \
        }                                                                    \
    } while (0)

    v8f acc0 = {}, acc1 = {}, acc2 = {};

    // prologue: chunks 0 and 1 in flight (7 async ops each per wave)
    ISSUE_CHUNK(0, 0);
    ISSUE_CHUNK(128, 1);

    #pragma unroll 1
    for (int c = 0; c < NCHUNK; ++c) {
        // chunk c retired when <=7 newer (chunk c+1) ops remain in this wave
        if (c < NCHUNK - 1) asm volatile("s_wait_asynccnt 0x7" ::: "memory");
        else                asm volatile("s_wait_asynccnt 0x0" ::: "memory");
        __syncthreads();    // all waves' copies of chunk c visible

        const _Float16* as = a_stage + (c & 1) * (A_ROWS * STG);
        const _Float16* bs = b_stage + (c & 1) * (B_ROWS * STG);
        const int brow = wv * 16 + l16;     // word column owned by this lane

        #pragma unroll
        for (int kk = 0; kk < 64; kk += 32) {
            // B fragment: chunk0 at kk+8*half, chunk1 at kk+16+8*half
            int ob = kk + 8 * half;
            v8h blo = *(const v8h*)(&bs[brow * STG + ob]);
            v8h bhi = *(const v8h*)(&bs[brow * STG + ob + 16]);
            v16h bfrag = __builtin_shufflevector(blo, bhi,
                0,1,2,3,4,5,6,7,8,9,10,11,12,13,14,15);
            // A fragments: chunk0 at kk+8*half, chunk1 at kk+16+8*half
            int o0 = kk + 8 * half, o1 = kk + 16 + 8 * half;
            v8h a0l = *(const v8h*)(&as[(0 * 16 + l16) * STG + o0]);
            v8h a0h = *(const v8h*)(&as[(0 * 16 + l16) * STG + o1]);
            v8h a1l = *(const v8h*)(&as[(1 * 16 + l16) * STG + o0]);
            v8h a1h = *(const v8h*)(&as[(1 * 16 + l16) * STG + o1]);
            v8h a2l = *(const v8h*)(&as[(2 * 16 + l16) * STG + o0]);
            v8h a2h = *(const v8h*)(&as[(2 * 16 + l16) * STG + o1]);
            v16h af0 = __builtin_shufflevector(a0l, a0h,
                0,1,2,3,4,5,6,7,8,9,10,11,12,13,14,15);
            v16h af1 = __builtin_shufflevector(a1l, a1h,
                0,1,2,3,4,5,6,7,8,9,10,11,12,13,14,15);
            v16h af2 = __builtin_shufflevector(a2l, a2h,
                0,1,2,3,4,5,6,7,8,9,10,11,12,13,14,15);
            acc0 = __builtin_amdgcn_wmma_f32_16x16x32_f16(
                false, af0, false, bfrag, (short)0, acc0, false, false);
            acc1 = __builtin_amdgcn_wmma_f32_16x16x32_f16(
                false, af1, false, bfrag, (short)0, acc1, false, false);
            acc2 = __builtin_amdgcn_wmma_f32_16x16x32_f16(
                false, af2, false, bfrag, (short)0, acc2, false, false);
        }
        __syncthreads();    // all waves done reading buf (c&1)
        if (c + 2 < NCHUNK) ISSUE_CHUNK((c + 2) * 128, (c & 1));
    }
#undef ISSUE_CHUNK
    __syncthreads();

    // scatter accumulators: D element v of lane -> row = tile*16 + v + 8*half, col = 16*wv + l16
    {
        const int c = wv * 16 + l16;
        #pragma unroll
        for (int v = 0; v < 8; ++v) {
            int r0 = v + 8 * half;            // 0..15  (<36 always)
            A_lds[r0 * 65 + c] = acc0[v];
            int r1 = 16 + v + 8 * half;       // 16..31 (<36 always)
            A_lds[r1 * 65 + c] = acc1[v];
            int r2 = 32 + v + 8 * half;       // 32..47 (keep <36)
            if (r2 < LI) A_lds[r2 * 65 + c] = acc2[v];
        }
    }
    __syncthreads();

    // 1st softmax: over valid words, per region row (thread-per-row)
    if (tid < LI) {
        const int r = tid;
        float m = -3.4e38f;
        for (int w = 0; w < nw; ++w) m = fmaxf(m, A_lds[r * 65 + w]);
        float ssum = 0.f;
        for (int w = 0; w < nw; ++w) {
            float e = __expf(A_lds[r * 65 + w] - m);
            W_lds[r * 65 + w] = e;
            ssum += e;
        }
        float inv = 1.f / ssum;
        for (int w = 0; w < nw; ++w) W_lds[r * 65 + w] *= inv;
    }
    __syncthreads();

    // 2nd softmax over regions (lambda-sharpened) + num + wc_norm^2 (thread-per-word)
    float mycos = 0.f;
    if (tid < nw) {
        const int w = tid;
        float m2 = -3.4e38f;
        for (int r = 0; r < LI; ++r) m2 = fmaxf(m2, LAMBDA_SM * W_lds[r * 65 + w]);
        float s2 = 0.f;
        for (int r = 0; r < LI; ++r) {
            float e = __expf(LAMBDA_SM * W_lds[r * 65 + w] - m2);
            W_lds[r * 65 + w] = e;          // own column: safe in-place
            s2 += e;
        }
        float inv = 1.f / s2;
        float num = 0.f;
        for (int r = 0; r < LI; ++r) {
            float a = W_lds[r * 65 + w] * inv;
            W_lds[r * 65 + w] = a;          // attn
            num += a * A_lds[r * 65 + w];
        }
        float wc2 = 0.f;
        for (int r = 0; r < LI; ++r) {
            float dot = 0.f;
            for (int ss = 0; ss < LI; ++ss)
                dot += G_lds[r * LI + ss] * W_lds[ss * 65 + w];
            wc2 += W_lds[r * 65 + w] * dot;
        }
        float cap = fmaxf(capn[j * LW + w], EPSV);
        float wn  = fmaxf(sqrtf(wc2), EPSV);
        mycos = num / (cap * wn);
    }
    red[tid] = mycos;
    __syncthreads();
    for (int off = 64; off > 0; off >>= 1) {
        if (tid < off) red[tid] += red[tid + off];
        __syncthreads();
    }
    if (tid == 0) scores[i * NB + j] = red[0] / (float)nw;  // scores = sim.T
}

// ---------------------------------------------------------------- hinge / max-violation loss
__global__ __launch_bounds__(128)
void loss_kernel(const float* __restrict__ scores, float* __restrict__ out) {
    const int t = threadIdx.x;
    __shared__ float red[128];
    float diag_t = scores[t * NB + t];
    float rowmax = 0.f, colmax = 0.f;   // hinge >= 0, diag forced 0 -> start at 0
    for (int k = 0; k < NB; ++k) {
        if (k == t) continue;
        float srow = scores[t * NB + k];           // scores[i=t][j=k], uses diag[i=t]
        rowmax = fmaxf(rowmax, fmaxf(0.f, MARGIN + srow - diag_t));
        float scol = scores[k * NB + t];           // scores[i=k][j=t], uses diag[j=t]
        colmax = fmaxf(colmax, fmaxf(0.f, MARGIN + scol - diag_t));
    }
    red[t] = rowmax + colmax;
    __syncthreads();
    for (int off = 64; off > 0; off >>= 1) {
        if (t < off) red[t] += red[t + off];
        __syncthreads();
    }
    if (t == 0) out[0] = red[0];
}

// ---------------------------------------------------------------------------
extern "C" void kernel_launch(void* const* d_in, const int* in_sizes, int n_in,
                              void* d_out, int out_size, void* d_ws, size_t ws_size,
                              hipStream_t stream) {
    (void)in_sizes; (void)n_in; (void)out_size; (void)ws_size;
    const float* im  = (const float*)d_in[0];   // (128, 36, 1024) f32
    const float* s   = (const float*)d_in[1];   // (128, 50, 1024) f32
    const int*   s_l = (const int*)d_in[2];     // (128,) int32
    float* out = (float*)d_out;

    char* ws = (char*)d_ws;
    size_t off = 0;
    _Float16* imh = (_Float16*)(ws + off); off += (size_t)NB * LI * DD * 2;   // 9.44 MB
    _Float16* shh = (_Float16*)(ws + off); off += (size_t)NB * LW * DD * 2;   // 13.1 MB
    float* capn   = (float*)(ws + off);    off += (size_t)NB * LW * 4;        // 25.6 KB
    float* G      = (float*)(ws + off);    off += (size_t)NB * LI * LI * 4;   // 664 KB
    float* scores = (float*)(ws + off);    off += (size_t)NB * NB * 4;        // 65.5 KB

    const int n_im = NB * LI * DD;
    const int n_s  = NB * LW * DD;
    cvt_f16_kernel<<<(n_im + 255) / 256, 256, 0, stream>>>(im, imh, n_im);
    cvt_f16_kernel<<<(n_s  + 255) / 256, 256, 0, stream>>>(s,  shh, n_s);
    capnorm_kernel<<<(NB * LW + 255) / 256, 256, 0, stream>>>(s, capn);
    gram_kernel<<<NB, 256, 0, stream>>>(im, G);
    pair_kernel<<<dim3(NB, NB), 128, 0, stream>>>(imh, shh, capn, G, s_l, scores);
    loss_kernel<<<1, 128, 0, stream>>>(scores, out);
}
